// VectorQuantizer_90202903151140
// MI455X (gfx1250) — matrix operand order repaired and checked by reading, not compiled, not run on hardware
//
#include <hip/hip_runtime.h>

// ---------------- CDNA5 WMMA types ----------------
typedef __bf16  bf16_t;
typedef bf16_t  v16bf __attribute__((ext_vector_type(16)));
typedef float   v8f   __attribute__((ext_vector_type(8)));
typedef unsigned short u16x8 __attribute__((ext_vector_type(8)));

#define DIM     512
#define KCODES  1024
#define MB      64          // rows per workgroup
#define ZS      520         // padded LDS row stride (halves) for Z (conflict-free A loads)
#define ES      40          // padded LDS row stride (halves) for E stage
#define EBUF    (64 * ES)   // halves per E stage plane

__device__ __forceinline__ unsigned short bf16_rne(float x) {
    unsigned b = __float_as_uint(x);
    unsigned r = b + 0x7FFFu + ((b >> 16) & 1u);
    return (unsigned short)(r >> 16);
}

__device__ __forceinline__ void bf16_split(float x, unsigned short& hi, unsigned short& lo) {
    hi = bf16_rne(x);
    float hf = __uint_as_float(((unsigned)hi) << 16);
    lo = bf16_rne(x - hf);
}

__device__ __forceinline__ v16bf ld_frag2(const unsigned short* p0, const unsigned short* p1) {
    union { u16x8 h[2]; v16bf v; } u;
    u.h[0] = *(const u16x8*)p0;
    u.h[1] = *(const u16x8*)p1;
    return u.v;
}

__device__ __forceinline__ v8f wmma_bf16(v16bf a, v16bf b, v8f c) {
    return __builtin_amdgcn_wmma_f32_16x16x32_bf16(false, a, false, b, (short)0, c, false, false);
}

// f32 min-reduce across each 16-lane half (4 xor-shuffle steps), then only the
// winning lane(s) commit (distKey<<32 | code) via LDS u64 atomic-min (ds_min_u64).
__device__ __forceinline__ void commit_tile(v8f acc, float e2v, unsigned code,
                                            int rb, int lane, unsigned long long* best) {
    float d[8], m[8];
#pragma unroll
    for (int v = 0; v < 8; ++v) {
        d[v] = fmaf(-2.0f, acc[v], e2v);   // ||e||^2 - 2 z.e
        m[v] = d[v];
    }
#pragma unroll
    for (int mask = 1; mask < 16; mask <<= 1) {
#pragma unroll
        for (int v = 0; v < 8; ++v)
            m[v] = fminf(m[v], __shfl_xor(m[v], mask, 32));
    }
    const int rofs = rb + ((lane & 16) ? 8 : 0);
#pragma unroll
    for (int v = 0; v < 8; ++v) {
        if (d[v] == m[v]) {
            unsigned fk = __float_as_uint(d[v]);
            fk ^= (fk >> 31) ? 0xFFFFFFFFu : 0x80000000u;  // order-preserving map
            unsigned long long key = (((unsigned long long)fk) << 32)
                                   | (unsigned long long)code;
            atomicMin(&best[rofs + v], key);
        }
    }
}

// ---- prep: split codebook to bf16 hi/lo and compute ||e||^2, once ----
__global__ __launch_bounds__(256)
void vq_prep(const float* __restrict__ ew,
             unsigned short* __restrict__ ehg,
             unsigned short* __restrict__ elg,
             float* __restrict__ e2g) {
    const int c = blockIdx.x * blockDim.x + threadIdx.x;   // code 0..1023
    const float* er = ew + (size_t)c * DIM;
    unsigned short* eh = ehg + (size_t)c * DIM;
    unsigned short* el = elg + (size_t)c * DIM;
    float ss = 0.0f;
    for (int j = 0; j < DIM / 4; ++j) {
        float4 v = ((const float4*)er)[j];
        ss += v.x * v.x + v.y * v.y + v.z * v.z + v.w * v.w;
        union { unsigned short s[4]; uint2 u; } ph, pl;
        bf16_split(v.x, ph.s[0], pl.s[0]);
        bf16_split(v.y, ph.s[1], pl.s[1]);
        bf16_split(v.z, ph.s[2], pl.s[2]);
        bf16_split(v.w, ph.s[3], pl.s[3]);
        *(uint2*)(eh + j * 4) = ph.u;
        *(uint2*)(el + j * 4) = pl.u;
    }
    e2g[c] = ss;
}

// ---- main: fused split-precision bf16 WMMA distance GEMM + argmin + gather ----
__global__ __launch_bounds__(256)
void vq_main(const float* __restrict__ z,
             const float* __restrict__ ew,
             const unsigned short* __restrict__ ehg,
             const unsigned short* __restrict__ elg,
             const float* __restrict__ e2g,
             float* __restrict__ out) {
    __shared__ unsigned short Zh[MB * ZS];
    __shared__ unsigned short Zl[MB * ZS];
    __shared__ unsigned short Est[2][2][EBUF];   // [buf][hi/lo], double-buffered stage
    __shared__ unsigned long long best[MB];

    const int t    = threadIdx.x;
    const int lane = t & 31;
    const int w    = t >> 5;        // wave 0..7
    const int wn   = w & 3;         // code-tile column (16 codes)
    const int wm   = w >> 2;        // row group of 32
    const int rowblk = blockIdx.x * MB;
    const int cst = t >> 2;         // staging: code 0..63
    const int qst = t & 3;          // staging: 8-half chunk

    // ---- stage Z block (64 x 512 f32) as bf16 hi/lo into LDS (once) ----
    {
        const int r = t >> 2;
        const int q = t & 3;
        const float* zr = z + (size_t)(rowblk + r) * DIM + q * 128;
        unsigned short* zh = Zh + r * ZS + q * 128;
        unsigned short* zl = Zl + r * ZS + q * 128;
#pragma unroll
        for (int j = 0; j < 32; ++j) {
            float4 v = ((const float4*)zr)[j];
            union { unsigned short s[4]; uint2 u; } ph, pl;
            bf16_split(v.x, ph.s[0], pl.s[0]);
            bf16_split(v.y, ph.s[1], pl.s[1]);
            bf16_split(v.z, ph.s[2], pl.s[2]);
            bf16_split(v.w, ph.s[3], pl.s[3]);
            *(uint2*)(zh + j * 4) = ph.u;
            *(uint2*)(zl + j * 4) = pl.u;
        }
    }
    if (t < MB) best[t] = ~0ull;
    __syncthreads();

    // fragment offsets (invariant)
    const int boff = (wn * 16 + (lane & 15)) * ES + ((lane < 16) ? 0 : 8);
    const int a0   = (wm * 32 + (lane & 15)) * ZS + ((lane < 16) ? 0 : 8);
    const int a1   = a0 + 16 * ZS;

    for (int cb = 0; cb < KCODES / 64; ++cb) {
        v8f acc0 = {}, acc1 = {};

        // prologue: stage kc=0 into buffer 0 (pure copy: E already bf16 in L2)
        {
            const size_t g = (size_t)(cb * 64 + cst) * DIM + qst * 8;
            const int   l = cst * ES + qst * 8;
            *(uint4*)(&Est[0][0][l]) = *(const uint4*)(ehg + g);
            *(uint4*)(&Est[0][1][l]) = *(const uint4*)(elg + g);
        }
        __syncthreads();

        for (int kc = 0; kc < DIM / 32; ++kc) {
            const int buf = kc & 1;
            if (kc + 1 < DIM / 32) {   // stage next chunk into other buffer
                const size_t g = (size_t)(cb * 64 + cst) * DIM + (kc + 1) * 32 + qst * 8;
                const int   l = cst * ES + qst * 8;
                *(uint4*)(&Est[buf ^ 1][0][l]) = *(const uint4*)(ehg + g);
                *(uint4*)(&Est[buf ^ 1][1][l]) = *(const uint4*)(elg + g);
            }

            const unsigned short* EhL = &Est[buf][0][0];
            const unsigned short* ElL = &Est[buf][1][0];
            v16bf Bh = ld_frag2(EhL + boff, EhL + boff + 16);
            v16bf Bl = ld_frag2(ElL + boff, ElL + boff + 16);

            const int ak = kc * 32;
            v16bf Ah0 = ld_frag2(Zh + a0 + ak, Zh + a0 + ak + 16);
            v16bf Al0 = ld_frag2(Zl + a0 + ak, Zl + a0 + ak + 16);
            v16bf Ah1 = ld_frag2(Zh + a1 + ak, Zh + a1 + ak + 16);
            v16bf Al1 = ld_frag2(Zl + a1 + ak, Zl + a1 + ak + 16);

            // split-precision: zh*eh + zl*eh + zh*el
            acc0 = wmma_bf16(Ah0, Bh, acc0);
            acc0 = wmma_bf16(Al0, Bh, acc0);
            acc0 = wmma_bf16(Ah0, Bl, acc0);
            acc1 = wmma_bf16(Ah1, Bh, acc1);
            acc1 = wmma_bf16(Al1, Bh, acc1);
            acc1 = wmma_bf16(Ah1, Bl, acc1);

            __syncthreads();   // publish staged buffer / retire reads of current
        }

        const unsigned code = (unsigned)(cb * 64 + wn * 16 + (lane & 15));
        const float e2v = e2g[code];     // L2 hit
        commit_tile(acc0, e2v, code, wm * 32,      lane, best);
        commit_tile(acc1, e2v, code, wm * 32 + 16, lane, best);
    }
    __syncthreads();

    // ---- gather winning codebook rows (f32, L2-resident) to output ----
    {
        const int r = t >> 2;
        const int q = t & 3;
        const unsigned idx = (unsigned)(best[r] & 0xFFFFFFFFull);
        const float4* src = (const float4*)(ew + (size_t)idx * DIM + q * 128);
        float4* dst = (float4*)(out + (size_t)(rowblk + r) * DIM + q * 128);
#pragma unroll
        for (int j = 0; j < 32; ++j) dst[j] = src[j];
    }
}

extern "C" void kernel_launch(void* const* d_in, const int* in_sizes, int n_in,
                              void* d_out, int out_size, void* d_ws, size_t ws_size,
                              hipStream_t stream) {
    const float* z  = (const float*)d_in[0];          // [65536, 512] f32
    const float* ew = (const float*)d_in[1];          // [1024, 512]  f32
    float* out = (float*)d_out;

    // workspace layout: E_hi (1 MB) | E_lo (1 MB) | e2 (4 KB)
    unsigned short* ehg = (unsigned short*)d_ws;
    unsigned short* elg = (unsigned short*)((char*)d_ws + (size_t)KCODES * DIM * 2);
    float*          e2g = (float*)((char*)d_ws + (size_t)KCODES * DIM * 4);

    vq_prep<<<dim3(KCODES / 256), dim3(256), 0, stream>>>(ew, ehg, elg, e2g);

    const int nrows = in_sizes[0] / DIM;              // 65536
    vq_main<<<dim3(nrows / MB), dim3(256), 0, stream>>>(z, ew, ehg, elg, e2g, out);
}
